// VectorQuantizer_15977278341694
// MI455X (gfx1250) — matrix operand order repaired and checked by reading, not compiled, not run on hardware
//
#include <hip/hip_runtime.h>
#include <hip/hip_bf16.h>
#include <math.h>

typedef __attribute__((ext_vector_type(16))) _Float16 f16x16;
typedef __attribute__((ext_vector_type(2)))  _Float16 f16x2;
typedef __attribute__((ext_vector_type(8)))  float    f32x8;
typedef __attribute__((ext_vector_type(4)))  unsigned int u32x4;
typedef __attribute__((ext_vector_type(8)))  int      i32x8;
typedef __attribute__((ext_vector_type(4)))  int      i32x4;

#define NUM_EMB   1024
#define EMB_DIM   256
#define N_ROWS    262144          // 8*32*32*32
#define ROWS_PER_BLOCK 128        // 8 waves * 16 rows
#define CHUNK_CODES 64            // codebook codes staged in LDS per pass
#define CB_STRIDE  264            // padded f16 stride (528B, 16B aligned, bank-spread)

#if __has_builtin(__builtin_amdgcn_tensor_load_to_lds)
#define USE_TDM 1
#else
#define USE_TDM 0
#endif

// ---------------- Prep: codebook -> f16, bias = -0.5*||e||^2, zero accums ----
__global__ __launch_bounds__(256) void vq_prep(const float* __restrict__ cb,
                                               _Float16* __restrict__ cb16,
                                               float* __restrict__ bias,
                                               float* __restrict__ counts,
                                               float* __restrict__ sse) {
  __shared__ float red[256];
  const int k = blockIdx.x;      // code index 0..1023
  const int t = threadIdx.x;     // dim index 0..255
  float e = cb[k * EMB_DIM + t];
  cb16[k * EMB_DIM + t] = (_Float16)e;
  red[t] = e * e;
  __syncthreads();
  for (int s = 128; s > 0; s >>= 1) {
    if (t < s) red[t] += red[t + s];
    __syncthreads();
  }
  if (t == 0) {
    bias[k] = -0.5f * red[0];
    counts[k] = 0.0f;
    if (k == 0) *sse = 0.0f;
  }
}

// B 32x16 f16 fragment: lanes 0-15: VGPR j<4 -> K=2j,2j+1 ; j>=4 -> K=16+2(j-4); lanes 16-31: same +8
__device__ __forceinline__ f16x16 load_b_frag(const _Float16* brow, int s, int h) {
  f16x16 b;
#pragma unroll
  for (int j = 0; j < 8; ++j) {
    f16x2 p = *(const f16x2*)(brow + s * 32 + ((j < 4) ? (2 * j) : (16 + 2 * (j - 4))) + 8 * h);
    b[2 * j]     = p.x;
    b[2 * j + 1] = p.y;
  }
  return b;
}

// ---------------- Main: WMMA scores + argmax + gather/quantize --------------
__global__ __launch_bounds__(256) void vq_main(const float* __restrict__ in,
                                               const float* __restrict__ cb,     // fp32 codebook (exact gather)
                                               const _Float16* __restrict__ cb16,
                                               const float* __restrict__ bias,
                                               float* __restrict__ out,
                                               float* __restrict__ counts,
                                               float* __restrict__ sseG) {
  __shared__ _Float16 ldsCb[CHUNK_CODES * CB_STRIDE];  // ~33.8 KB
  __shared__ int   idxLds[ROWS_PER_BLOCK];
  __shared__ float red[256];

  const int tid  = threadIdx.x;
  const int lane = tid & 31;
  const int wave = tid >> 5;
  const int h    = lane >> 4;       // half-wave select
  const int r    = lane & 15;       // row within tile / column within tile
  const int blockRow = blockIdx.x * ROWS_PER_BLOCK;
  const int waveRow  = blockRow + wave * 16;

  // ---- Load A fragments for this wave's 16 rows: whole D=256 resident -----
  // 16-bit A 16x32 layout: lanes 0-15: VGPR j<4 -> K=2j,2j+1 ; j>=4 -> K=16+2(j-4)
  //                        lanes 16-31: same +8
  const float* rowPtr = in + (size_t)(waveRow + r) * EMB_DIM;
  f16x16 afr[8];
#pragma unroll
  for (int s = 0; s < 8; ++s) {
#pragma unroll
    for (int j = 0; j < 8; ++j) {
      int kl = ((j < 4) ? (2 * j) : (16 + 2 * (j - 4))) + 8 * h;
      float2 f = *(const float2*)(rowPtr + s * 32 + kl);
      afr[s][2 * j]     = (_Float16)f.x;
      afr[s][2 * j + 1] = (_Float16)f.y;
    }
  }

  float bestV[8];
  int   bestI[8];
#pragma unroll
  for (int v = 0; v < 8; ++v) { bestV[v] = -INFINITY; bestI[v] = 0; }

  // ---- Sweep all 1024 codes in LDS-staged chunks of 64 ---------------------
  for (int chunk = 0; chunk < NUM_EMB / CHUNK_CODES; ++chunk) {
    __syncthreads();   // all waves done reading previous chunk

#if USE_TDM
    // Tensor Data Mover: one descriptor DMAs the 64x256 f16 tile into LDS,
    // hardware-padding 4 DWORDs after every 128-DWORD row -> stride 264 f16.
    if (wave == 0) {
      unsigned long long gaddr =
          (unsigned long long)(const void*)cb16 +
          (unsigned long long)chunk * (CHUNK_CODES * EMB_DIM * 2);
      unsigned ldsBase = (unsigned)(unsigned long long)(void*)&ldsCb[0]; // LDS byte offset
      u32x4 g0;
      g0[0] = 1u;                                            // count=1, user descriptor
      g0[1] = ldsBase;                                       // lds_addr
      g0[2] = (unsigned)(gaddr & 0xFFFFFFFFull);             // global_addr[31:0]
      g0[3] = (unsigned)((gaddr >> 32) & 0x01FFFFFFull)      // global_addr[56:32]
            | (2u << 30);                                    // type=2 ("image")
      i32x8 g1;
      g1[0] = (1 << 16)    // data_size = 2 bytes
            | (1 << 20)    // pad_enable
            | (6 << 22)    // pad_interval: 128 DWORDs (one 512B row)
            | (3 << 25);   // pad_amount: 4 DWORDs (8 f16)
      g1[1] = (EMB_DIM & 0xFFFF) << 16;                      // tensor_dim0 lo
      g1[2] = (EMB_DIM >> 16) | ((NUM_EMB & 0xFFFF) << 16);  // dim0 hi | tensor_dim1 lo
      g1[3] = (NUM_EMB >> 16) | (EMB_DIM << 16);             // dim1 hi | tile_dim0=256
      g1[4] = CHUNK_CODES;                                   // tile_dim1=64, tile_dim2=0
      g1[5] = EMB_DIM;                                       // tensor_dim0_stride lo
      g1[6] = 0;
      g1[7] = 0;
      i32x4 z4 = {0, 0, 0, 0};
#if __clang_major__ >= 23
      i32x8 z8 = {0, 0, 0, 0, 0, 0, 0, 0};
      __builtin_amdgcn_tensor_load_to_lds(g0, g1, z4, z4, z8, 0);
#else
      __builtin_amdgcn_tensor_load_to_lds(g0, g1, z4, z4, 0);
#endif
#if __has_builtin(__builtin_amdgcn_s_wait_tensorcnt)
      __builtin_amdgcn_s_wait_tensorcnt(0);
#else
      asm volatile("s_wait_tensorcnt 0x0" ::: "memory");
#endif
    }
#else
    // Fallback cooperative copy: 64 codes * 256 f16 = 32KB, 16B vector moves
    {
      const uint4* src = (const uint4*)(cb16 + (size_t)chunk * CHUNK_CODES * EMB_DIM);
      for (int i = tid; i < CHUNK_CODES * 32; i += 256) {
        int cr = i >> 5, seg = i & 31;
        uint4 v = src[cr * 32 + seg];
        *(uint4*)(&ldsCb[cr * CB_STRIDE + seg * 8]) = v;
      }
    }
#endif

    // Preload this chunk's bias values (hidden behind the LDS fill + barrier)
    float biasReg[4];
#pragma unroll
    for (int kt = 0; kt < 4; ++kt)
      biasReg[kt] = bias[chunk * CHUNK_CODES + kt * 16 + r];

    __syncthreads();   // chunk visible to all waves

#pragma unroll
    for (int kt = 0; kt < CHUNK_CODES / 16; ++kt) {
      const int code = chunk * CHUNK_CODES + kt * 16 + r;  // this lane's column code
      f32x8 c;
#pragma unroll
      for (int v = 0; v < 8; ++v) c[v] = biasReg[kt];  // C init = -0.5*||e||^2

      const _Float16* brow = &ldsCb[(kt * 16 + r) * CB_STRIDE];
      // Software-pipelined B: issue ds_loads for s+1 before the WMMA on s
      f16x16 bcur = load_b_frag(brow, 0, h);
      f16x16 bnext = bcur;
#pragma unroll
      for (int s = 0; s < 8; ++s) {
        if (s < 7) bnext = load_b_frag(brow, s + 1, h);
        c = __builtin_amdgcn_wmma_f32_16x16x32_f16(
                false, afr[s], false, bcur, (short)0, c, false, false);
        bcur = bnext;
      }
#pragma unroll
      for (int v = 0; v < 8; ++v) {
        if (c[v] > bestV[v]) { bestV[v] = c[v]; bestI[v] = code; }
      }
    }
  }

  // ---- Argmax reduction across the 16 lanes sharing each row ---------------
#pragma unroll
  for (int off = 8; off >= 1; off >>= 1) {
#pragma unroll
    for (int v = 0; v < 8; ++v) {
      float ov = __shfl_xor(bestV[v], off, 32);
      int   oi = __shfl_xor(bestI[v], off, 32);
      if (ov > bestV[v] || (ov == bestV[v] && oi < bestI[v])) {
        bestV[v] = ov; bestI[v] = oi;
      }
    }
  }
  if (lane == 0) {
#pragma unroll
    for (int v = 0; v < 8; ++v) idxLds[wave * 16 + v] = bestI[v];        // rows 0..7
  } else if (lane == 16) {
#pragma unroll
    for (int v = 0; v < 8; ++v) idxLds[wave * 16 + 8 + v] = bestI[v];    // rows 8..15
  }
  __syncthreads();

  // ---- Histogram for perplexity --------------------------------------------
  if (tid < ROWS_PER_BLOCK) atomicAdd(&counts[idxLds[tid]], 1.0f);

  // ---- Gather fp32 codebook rows (exact), write coalesced, accumulate SSE --
  float sse = 0.0f;
  for (int row = 0; row < ROWS_PER_BLOCK; ++row) {
    const int code = idxLds[row];
    const size_t o = (size_t)(blockRow + row) * EMB_DIM + tid;
    float q = cb[(size_t)code * EMB_DIM + tid];
    float x = in[o];
    float d = q - x;
    sse += d * d;
    out[o] = q;   // straight-through output == quantized value
  }
  red[tid] = sse;
  __syncthreads();
  for (int s = 128; s > 0; s >>= 1) {
    if (tid < s) red[tid] += red[tid + s];
    __syncthreads();
  }
  if (tid == 0) atomicAdd(sseG, red[0]);
}

// ---------------- Finalize: loss + perplexity -------------------------------
__global__ __launch_bounds__(256) void vq_finalize(const float* __restrict__ counts,
                                                   const float* __restrict__ sseG,
                                                   float* __restrict__ tail) {
  __shared__ float red[256];
  const int t = threadIdx.x;
  float hacc = 0.0f;
  for (int k = t; k < NUM_EMB; k += 256) {
    float avg = counts[k] * (1.0f / (float)N_ROWS);
    hacc += avg * logf(avg + 1e-10f);
  }
  red[t] = hacc;
  __syncthreads();
  for (int s = 128; s > 0; s >>= 1) {
    if (t < s) red[t] += red[t + s];
    __syncthreads();
  }
  if (t == 0) {
    // q_latent_loss == e_latent_loss numerically -> loss = 1.25 * MSE
    tail[0] = 1.25f * (*sseG) / ((float)N_ROWS * (float)EMB_DIM);
    tail[1] = expf(-red[0]);
  }
}

extern "C" void kernel_launch(void* const* d_in, const int* in_sizes, int n_in,
                              void* d_out, int out_size, void* d_ws, size_t ws_size,
                              hipStream_t stream) {
  (void)in_sizes; (void)n_in; (void)out_size; (void)ws_size;
  const float* inputs   = (const float*)d_in[0];   // [262144, 256] f32
  const float* codebook = (const float*)d_in[1];   // [1024, 256] f32
  float* out = (float*)d_out;                      // quantized (67108864) + loss + perplexity

  char* ws = (char*)d_ws;
  _Float16* cb16 = (_Float16*)(ws);                               // 512 KB
  float* bias    = (float*)(ws + 524288);                         // 4 KB
  float* counts  = (float*)(ws + 524288 + 4096);                  // 4 KB
  float* sse     = (float*)(ws + 524288 + 8192);                  // 4 B

  vq_prep<<<NUM_EMB, 256, 0, stream>>>(codebook, cb16, bias, counts, sse);
  vq_main<<<N_ROWS / ROWS_PER_BLOCK, 256, 0, stream>>>(
      inputs, codebook, cb16, bias, out, counts, sse);
  vq_finalize<<<1, 256, 0, stream>>>(counts, sse, out + (size_t)N_ROWS * EMB_DIM);
}
